// QuotRemLinear_15118284882234
// MI455X (gfx1250) — matrix operand order repaired and checked by reading, not compiled, not run on hardware
//
#include <hip/hip_runtime.h>
#include <cstdint>

typedef __bf16 bf16;
typedef __attribute__((ext_vector_type(4)))  bf16  bf16x4;
typedef __attribute__((ext_vector_type(8)))  bf16  bf16x8;
typedef __attribute__((ext_vector_type(16))) bf16  bf16x16;
typedef __attribute__((ext_vector_type(8)))  float floatx8;

#define M_TOT 8192      // B*S = 4*2048
#define K_TOT 4096
#define N_TOT 11008
#define BM 256
#define BN 128
#define BK 64
#define LDSS 72         // padded LDS row stride (bf16): 144B rows -> conflict-free 16B frag loads
#define NBUF 3
#define A_ELEMS (BM * LDSS)
#define B_ELEMS (BN * LDSS)
#define BUF_ELEMS (A_ELEMS + B_ELEMS)   // per-buffer bf16 elems (384*72 = 27648 -> 55296 B)

// ---------------------------------------------------------------------------
// Kernel 1: quotient-remainder activation fake-quant -> bf16
// One wave (32 lanes) per group of 128; lane handles 4 contiguous floats.
// ---------------------------------------------------------------------------
__global__ __launch_bounds__(256)
void qr_recon_bf16_kernel(const float* __restrict__ x, bf16* __restrict__ out) {
    const int  lane = threadIdx.x & 31;
    const long long wave = ((long long)blockIdx.x * blockDim.x + threadIdx.x) >> 5;
    const long long idx  = wave * 128 + (long long)lane * 4;

    float4 v = *(const float4*)(x + idx);

    float m = fmaxf(fmaxf(fabsf(v.x), fabsf(v.y)), fmaxf(fabsf(v.z), fabsf(v.w)));
#pragma unroll
    for (int off = 16; off > 0; off >>= 1) m = fmaxf(m, __shfl_xor(m, off, 32));
    m = fmaxf(m, 1e-8f);

    float t = m * (1.0f / 7.0f);
    float base = 64.0f;
    if (t <= 32.0f) base = 32.0f;
    if (t <= 16.0f) base = 16.0f;
    if (t <= 8.0f)  base = 8.0f;
    if (t <= 4.0f)  base = 4.0f;
    if (t <= 2.0f)  base = 2.0f;
    const float ib = 1.0f / base;   // exact (pow2)

    float qx = fminf(fmaxf(rintf(v.x * ib), -7.0f), 7.0f);
    float qy = fminf(fmaxf(rintf(v.y * ib), -7.0f), 7.0f);
    float qz = fminf(fmaxf(rintf(v.z * ib), -7.0f), 7.0f);
    float qw = fminf(fmaxf(rintf(v.w * ib), -7.0f), 7.0f);

    float rx = v.x - base * qx;
    float ry = v.y - base * qy;
    float rz = v.z - base * qz;
    float rw = v.w - base * qw;

    float mr = fmaxf(fmaxf(fabsf(rx), fabsf(ry)), fmaxf(fabsf(rz), fabsf(rw)));
#pragma unroll
    for (int off = 16; off > 0; off >>= 1) mr = fmaxf(mr, __shfl_xor(mr, off, 32));
    mr = fmaxf(mr, 1e-8f);
    const float sr = mr * (1.0f / 7.0f);

    float dx = fminf(fmaxf(rintf(rx / sr), -8.0f), 7.0f) * sr;
    float dy = fminf(fmaxf(rintf(ry / sr), -8.0f), 7.0f) * sr;
    float dz = fminf(fmaxf(rintf(rz / sr), -8.0f), 7.0f) * sr;
    float dw = fminf(fmaxf(rintf(rw / sr), -8.0f), 7.0f) * sr;

    bf16x4 o;
    o.x = (bf16)(base * qx + dx);
    o.y = (bf16)(base * qy + dy);
    o.z = (bf16)(base * qz + dz);
    o.w = (bf16)(base * qw + dw);
    *(bf16x4*)(out + idx) = o;
}

// ---------------------------------------------------------------------------
// Kernel 2: weight fp32 -> bf16 (streaming convert)
// ---------------------------------------------------------------------------
__global__ __launch_bounds__(256)
void w_cvt_bf16_kernel(const float* __restrict__ w, bf16* __restrict__ out) {
    const long long i = ((long long)blockIdx.x * blockDim.x + threadIdx.x) * 4;
    float4 v = *(const float4*)(w + i);
    bf16x4 o;
    o.x = (bf16)v.x; o.y = (bf16)v.y; o.z = (bf16)v.z; o.w = (bf16)v.w;
    *(bf16x4*)(out + i) = o;
}

// ---------------------------------------------------------------------------
// Kernel 3: C[M,N] = A[M,K] * B[N,K]^T + bias, bf16 inputs, f32 accum (WMMA)
// 256 threads = 8 waves; block tile 256x128, K-tile 64.
// Wave grid 4(M) x 2(N): wave tile 64x64 = 4x4 fragments of 16x16
//   -> 16 ds_load_b128 per 16 WMMA (A and B frags each reused 4x).
// Triple-buffered LDS fed by async global->LDS b128 copies (ASYNCcnt),
// 2 K-tiles ahead; s_wait_asynccnt 12 retires tile kt while kt+1 flies.
// ---------------------------------------------------------------------------
__global__ __launch_bounds__(256)
void gemm_bf16_wmma_kernel(const bf16* __restrict__ A, const bf16* __restrict__ B,
                           const float* __restrict__ bias, float* __restrict__ C) {
    __shared__ __align__(16) bf16 smem[NBUF * BUF_ELEMS];   // 3 x 55296 B = 162 KB

    const int tid    = threadIdx.x;
    const int lane   = tid & 31;
    const int waveId = tid >> 5;
    const int wm = waveId >> 1;  // 0..3 : 64 rows each (within BM=256)
    const int wn = waveId & 1;   // 0..1 : 64 cols each (within BN=128)
    const int bm0 = blockIdx.y * BM;
    const int bn0 = blockIdx.x * BN;

    // gfx1250: generic address of LDS object = {shared aperture, lds byte offset}
    const unsigned lds_base = (unsigned)(uintptr_t)(void*)smem;

    floatx8 acc[4][4];
#pragma unroll
    for (int i = 0; i < 4; ++i)
#pragma unroll
        for (int j = 0; j < 4; ++j) acc[i][j] = floatx8{};

    // Issue 12 async 16B copies (8 A-chunks + 4 B-chunks) for K-tile kt.
    auto issueAsync = [&](int kt) {
        const long long k0 = (long long)kt * BK;
        const unsigned boff = lds_base + (unsigned)((kt % NBUF) * BUF_ELEMS) * 2u;
#pragma unroll
        for (int p = 0; p < 8; ++p) {       // A: 2048 chunks of 8 bf16
            const int c   = tid + p * 256;
            const int row = c >> 3;         // 0..255
            const int col = (c & 7) << 3;   // 0..56
            const unsigned la = boff + (unsigned)(row * LDSS + col) * 2u;
            const unsigned long long ga =
                (unsigned long long)(uintptr_t)(A + (long long)(bm0 + row) * K_TOT + k0 + col);
            asm volatile("global_load_async_to_lds_b128 %0, %1, off"
                         :: "v"(la), "v"(ga) : "memory");
        }
#pragma unroll
        for (int p = 0; p < 4; ++p) {       // B: 1024 chunks of 8 bf16
            const int c   = tid + p * 256;
            const int row = c >> 3;         // 0..127
            const int col = (c & 7) << 3;
            const unsigned lb = boff + (unsigned)(A_ELEMS + row * LDSS + col) * 2u;
            const unsigned long long gb =
                (unsigned long long)(uintptr_t)(B + (long long)(bn0 + row) * K_TOT + k0 + col);
            asm volatile("global_load_async_to_lds_b128 %0, %1, off"
                         :: "v"(lb), "v"(gb) : "memory");
        }
    };

    issueAsync(0);
    issueAsync(1);

    const int r15   = lane & 15;
    const int klo_a = (lane >> 4) << 3;   // A frag: lanes>=16 -> K offset +8 (halves at K, K+16)
    const int klo_b = (lane >> 4) << 3;   // B frag: lanes>=16 -> K offset +8 (halves at K, K+16)
    const int NT = K_TOT / BK;            // 64

    for (int kt = 0; kt < NT; ++kt) {
        // Retire my tile-kt batch: in-order completion => suffix of <=12 outstanding
        // async ops lies entirely in tile kt+1's batch. Last tile: drain fully.
        if (kt + 1 < NT) asm volatile("s_wait_asynccnt 12" ::: "memory");
        else             asm volatile("s_wait_asynccnt 0" ::: "memory");
        __syncthreads();                   // all waves' tile-kt data visible; frees buf (kt+2)%3
        if (kt + 2 < NT) issueAsync(kt + 2);

        const bf16* sa = smem + (kt % NBUF) * BUF_ELEMS;
        const bf16* sb = sa + A_ELEMS;

#pragma unroll
        for (int kk = 0; kk < BK; kk += 32) {
            bf16x16 af[4], bfrag[4];
#pragma unroll
            for (int i = 0; i < 4; ++i) {
                const bf16* p = sa + (wm * 64 + i * 16 + r15) * LDSS + kk + klo_a;
                bf16x8 lo = *(const bf16x8*)p;          // K = klo_a .. +7
                bf16x8 hi = *(const bf16x8*)(p + 16);   // K = klo_a+16 .. +23
                af[i] = __builtin_shufflevector(lo, hi, 0,1,2,3,4,5,6,7,8,9,10,11,12,13,14,15);
            }
#pragma unroll
            for (int j = 0; j < 4; ++j) {
                const bf16* p = sb + (wn * 64 + j * 16 + r15) * LDSS + kk + klo_b;
                bf16x8 lo = *(const bf16x8*)p;          // K = klo_b .. +7
                bf16x8 hi = *(const bf16x8*)(p + 16);   // K = klo_b+16 .. +23
                bfrag[j] = __builtin_shufflevector(lo, hi, 0,1,2,3,4,5,6,7,8,9,10,11,12,13,14,15);
            }
#pragma unroll
            for (int i = 0; i < 4; ++i)
#pragma unroll
                for (int j = 0; j < 4; ++j)
                    acc[i][j] = __builtin_amdgcn_wmma_f32_16x16x32_bf16(
                        false, af[i], false, bfrag[j], (short)0, acc[i][j], false, false);
        }
    }

    // Epilogue: C/D layout -> lane (0..15, N=lane), VGPR v = row v (+8 for lanes>=16)
    const int mhalf = (lane >> 4) << 3;
#pragma unroll
    for (int i = 0; i < 4; ++i) {
#pragma unroll
        for (int j = 0; j < 4; ++j) {
            const int n     = bn0 + wn * 64 + j * 16 + r15;
            const int mbase = bm0 + wm * 64 + i * 16 + mhalf;
            const float bv  = bias[n];
#pragma unroll
            for (int v = 0; v < 8; ++v)
                C[(long long)(mbase + v) * N_TOT + n] = acc[i][j][v] + bv;
        }
    }
}

// ---------------------------------------------------------------------------
extern "C" void kernel_launch(void* const* d_in, const int* in_sizes, int n_in,
                              void* d_out, int out_size, void* d_ws, size_t ws_size,
                              hipStream_t stream) {
    const float* x    = (const float*)d_in[0];  // [4,2048,4096]
    const float* w    = (const float*)d_in[1];  // [11008,4096]
    const float* bias = (const float*)d_in[2];  // [11008]
    float*       out  = (float*)d_out;          // [4,2048,11008]

    bf16* A = (bf16*)d_ws;                            // 8192*4096 bf16 = 64 MB
    bf16* W = A + (size_t)M_TOT * K_TOT;              // 11008*4096 bf16 = 86 MB

    // 1) activation fake-quant -> bf16 : 262144 groups, 1 wave each, 8 waves/block
    qr_recon_bf16_kernel<<<(M_TOT * (K_TOT / 128)) / 8, 256, 0, stream>>>(x, A);

    // 2) weight convert: 45,088,768 elems / 4 per thread / 256 = 44032 blocks
    w_cvt_bf16_kernel<<<(N_TOT * K_TOT) / 4 / 256, 256, 0, stream>>>(w, W);

    // 3) WMMA GEMM: grid (N/128, M/256) = (86, 32)
    dim3 grid(N_TOT / BN, M_TOT / BM);
    gemm_bf16_wmma_kernel<<<grid, 256, 0, stream>>>(A, W, bias, out);
}
